// Qwen2MoeSparseMoeBlock_72894184948087
// MI455X (gfx1250) — compile-verified, matrix operand order checked by reading, not run on hardware
//
#include <hip/hip_runtime.h>
#include <hip/hip_bf16.h>

typedef __bf16 bf16;
typedef __attribute__((ext_vector_type(16))) __bf16 v16bf;
typedef __attribute__((ext_vector_type(8)))  float  v8f;

#define T_TOK 4096
#define HID   2048
#define NEXP  8
#define FINT  1408
#define CAP   2048   // per-expert token capacity (expected ~1024)
#define KC    32     // K step per WMMA (bf16 16x16x32)
#define BM    128
#define BN    128

// ---- CDNA5 async global->LDS DMA (ASYNCcnt-tracked), per ISA 08_async_tensor ----
__device__ __forceinline__ void async_ld_b128(unsigned int lds_off, const bf16* gaddr) {
    asm volatile("global_load_async_to_lds_b128 %0, %1, off"
                 :: "v"(lds_off), "v"(gaddr)
                 : "memory");
}
__device__ __forceinline__ void wait_async0() {
    asm volatile("s_wait_asynccnt 0" ::: "memory");
}

// ---------------- zero out + counters ----------------
__global__ void moe_zero_kernel(float* __restrict__ out, long n, int* __restrict__ counts) {
    long i = (long)blockIdx.x * blockDim.x + threadIdx.x;
    if (i < n) out[i] = 0.f;
    if (blockIdx.x == 0 && threadIdx.x < NEXP) counts[threadIdx.x] = 0;
}

// ---------------- fp32 -> bf16 convert ----------------
__global__ void moe_cvt_bf16_kernel(const float* __restrict__ src, bf16* __restrict__ dst, long n) {
    long i = ((long)blockIdx.x * blockDim.x + threadIdx.x) * 4;
    if (i + 3 < n) {
        float4 v = *(const float4*)(src + i);
        dst[i + 0] = (bf16)v.x; dst[i + 1] = (bf16)v.y;
        dst[i + 2] = (bf16)v.z; dst[i + 3] = (bf16)v.w;
    } else {
        for (long j = i; j < n; ++j) dst[j] = (bf16)src[j];
    }
}

// ---------------- router: softmax + top-2, build expert groups ----------------
__global__ void moe_router_kernel(const float* __restrict__ x, const float* __restrict__ gw,
                                  int* __restrict__ counts, int* __restrict__ tok_list,
                                  float* __restrict__ wts_list) {
    int wid  = threadIdx.x >> 5;
    int lane = threadIdx.x & 31;
    int t = blockIdx.x * 8 + wid;     // one wave32 per token
    if (t >= T_TOK) return;
    float acc[NEXP];
#pragma unroll
    for (int e = 0; e < NEXP; ++e) acc[e] = 0.f;
    const float* xr = x + (size_t)t * HID;
    for (int h = lane; h < HID; h += 32) {
        float xv = xr[h];
        const float4* g = (const float4*)(gw + (size_t)h * NEXP);
        float4 g0 = g[0], g1 = g[1];
        acc[0] += xv * g0.x; acc[1] += xv * g0.y; acc[2] += xv * g0.z; acc[3] += xv * g0.w;
        acc[4] += xv * g1.x; acc[5] += xv * g1.y; acc[6] += xv * g1.z; acc[7] += xv * g1.w;
    }
#pragma unroll
    for (int e = 0; e < NEXP; ++e)
#pragma unroll
        for (int off = 16; off; off >>= 1) acc[e] += __shfl_xor(acc[e], off, 32);
    if (lane == 0) {
        float m = acc[0];
#pragma unroll
        for (int e = 1; e < NEXP; ++e) m = fmaxf(m, acc[e]);
        float p[NEXP], s = 0.f;
#pragma unroll
        for (int e = 0; e < NEXP; ++e) { p[e] = __expf(acc[e] - m); s += p[e]; }
        float inv = 1.f / s;
        int e1 = 0; float p1 = -1.f;
#pragma unroll
        for (int e = 0; e < NEXP; ++e) if (p[e] > p1) { p1 = p[e]; e1 = e; }
        int e2 = 0; float p2 = -1.f;
#pragma unroll
        for (int e = 0; e < NEXP; ++e) if (e != e1 && p[e] > p2) { p2 = p[e]; e2 = e; }
        int s1 = atomicAdd(&counts[e1], 1);
        if (s1 < CAP) { tok_list[e1 * CAP + s1] = t; wts_list[e1 * CAP + s1] = p1 * inv; }
        int s2 = atomicAdd(&counts[e2], 1);
        if (s2 < CAP) { tok_list[e2 * CAP + s2] = t; wts_list[e2 * CAP + s2] = p2 * inv; }
    }
}

// ---------------- grouped GEMM: gate & up fused, SiLU epilogue ----------------
__launch_bounds__(256)
__global__ void moe_gateup_kernel(const bf16* __restrict__ xb,
                                  const bf16* __restrict__ wg,
                                  const bf16* __restrict__ wu,
                                  const int* __restrict__ counts,
                                  const int* __restrict__ tok_list,
                                  bf16* __restrict__ hbuf) {
    __shared__ bf16 lA [2][BM * KC];   // [M][K], filled by async DMA
    __shared__ bf16 lBg[2][BN * KC];   // [N][K] (transposed at fill)
    __shared__ bf16 lBu[2][BN * KC];
    __shared__ int  toks[BM];

    int e = blockIdx.z;
    int cnt = counts[e]; if (cnt > CAP) cnt = CAP;
    int mbase = blockIdx.y * BM;
    if (mbase >= cnt) return;
    int nbase = blockIdx.x * BN;
    int tid = threadIdx.x, lane = tid & 31, wid = tid >> 5;

    if (tid < BM) {
        int r = mbase + tid;
        toks[tid] = (r < cnt) ? tok_list[e * CAP + r] : 0;
    }
    __syncthreads();

    const bf16* wgE = wg + (size_t)e * HID * FINT + nbase;
    const bf16* wuE = wu + (size_t)e * HID * FINT + nbase;

    int waveM = wid >> 2;          // 0..1 -> 64 rows each
    int waveN = wid & 3;           // 0..3 -> 32 cols each
    v8f zero = {0.f,0.f,0.f,0.f,0.f,0.f,0.f,0.f};
    v8f accG[4][2], accU[4][2];
#pragma unroll
    for (int i = 0; i < 4; ++i)
#pragma unroll
        for (int j = 0; j < 2; ++j) { accG[i][j] = zero; accU[i][j] = zero; }

    // A staging: 128 rows x 32 cols, 16 bf16 (32B) per thread -> 2 async b128
    int arow = tid >> 1;
    int acol = (tid & 1) * 16;
    const bf16* xrow = xb + (size_t)toks[arow] * HID + acol;
    unsigned int aoff[2];
    aoff[0] = (unsigned int)(size_t)&lA[0][arow * KC + acol];
    aoff[1] = (unsigned int)(size_t)&lA[1][arow * KC + acol];

    // B staging: 16 bf16 per thread, 2 chunks of 8 along f
    int lin0 = tid * 8,          kk0 = lin0 >> 7, ff0 = lin0 & 127;
    int lin1 = (tid + 256) * 8,  kk1 = lin1 >> 7, ff1 = lin1 & 127;

#define STAGE_B(K0, BUF)                                                          \
    {                                                                             \
        uint4 vg0 = *(const uint4*)(wgE + (size_t)((K0) + kk0) * FINT + ff0);     \
        uint4 vu0 = *(const uint4*)(wuE + (size_t)((K0) + kk0) * FINT + ff0);     \
        uint4 vg1 = *(const uint4*)(wgE + (size_t)((K0) + kk1) * FINT + ff1);     \
        uint4 vu1 = *(const uint4*)(wuE + (size_t)((K0) + kk1) * FINT + ff1);     \
        const bf16* pg0 = (const bf16*)&vg0; const bf16* pu0 = (const bf16*)&vu0; \
        const bf16* pg1 = (const bf16*)&vg1; const bf16* pu1 = (const bf16*)&vu1; \
        _Pragma("unroll")                                                         \
        for (int j = 0; j < 8; ++j) {                                             \
            lBg[BUF][(ff0 + j) * KC + kk0] = pg0[j];                              \
            lBu[BUF][(ff0 + j) * KC + kk0] = pu0[j];                              \
            lBg[BUF][(ff1 + j) * KC + kk1] = pg1[j];                              \
            lBu[BUF][(ff1 + j) * KC + kk1] = pu1[j];                              \
        }                                                                         \
    }

    // prologue: stage tile 0 into buffer 0
    async_ld_b128(aoff[0],      xrow);
    async_ld_b128(aoff[0] + 16, xrow + 8);
    STAGE_B(0, 0)
    wait_async0();
    __syncthreads();

    for (int k0 = 0, it = 0; k0 < HID; k0 += KC, ++it) {
        int cur = it & 1, nxt = cur ^ 1;
        bool more = (k0 + KC) < HID;
        if (more) {
            async_ld_b128(aoff[nxt],      xrow + k0 + KC);
            async_ld_b128(aoff[nxt] + 16, xrow + k0 + KC + 8);
            STAGE_B(k0 + KC, nxt)
        }
        int mrow  = lane & 15;
        int khalf = (lane >> 4) * 16;
        v16bf a[4], bgf[2], buf_[2];
#pragma unroll
        for (int mi = 0; mi < 4; ++mi)
            a[mi] = *(const v16bf*)&lA[cur][(waveM * 64 + mi * 16 + mrow) * KC + khalf];
#pragma unroll
        for (int ni = 0; ni < 2; ++ni) {
            bgf[ni]  = *(const v16bf*)&lBg[cur][(waveN * 32 + ni * 16 + mrow) * KC + khalf];
            buf_[ni] = *(const v16bf*)&lBu[cur][(waveN * 32 + ni * 16 + mrow) * KC + khalf];
        }
#pragma unroll
        for (int mi = 0; mi < 4; ++mi)
#pragma unroll
            for (int ni = 0; ni < 2; ++ni) {
                accG[mi][ni] = __builtin_amdgcn_wmma_f32_16x16x32_bf16(
                    false, a[mi], false, bgf[ni], (short)0, accG[mi][ni], false, false);
                accU[mi][ni] = __builtin_amdgcn_wmma_f32_16x16x32_bf16(
                    false, a[mi], false, buf_[ni], (short)0, accU[mi][ni], false, false);
            }
        if (more) wait_async0();
        __syncthreads();
    }
#undef STAGE_B

    // epilogue: h = silu(g) * u  -> hbuf (bf16), per-expert compacted rows
    int ncol  = lane & 15;
    int mhalf = (lane >> 4) * 8;
#pragma unroll
    for (int mi = 0; mi < 4; ++mi)
#pragma unroll
        for (int ni = 0; ni < 2; ++ni)
#pragma unroll
            for (int v = 0; v < 8; ++v) {
                int m = waveM * 64 + mi * 16 + mhalf + v;
                int n = nbase + waveN * 32 + ni * 16 + ncol;
                if (mbase + m < cnt) {
                    float g = accG[mi][ni][v];
                    float u = accU[mi][ni][v];
                    float h = (g / (1.f + __expf(-g))) * u;
                    hbuf[((size_t)e * CAP + mbase + m) * FINT + n] = (bf16)h;
                }
            }
}

// ---------------- grouped GEMM: down, scaled atomic scatter-add ----------------
__launch_bounds__(256)
__global__ void moe_down_kernel(const bf16* __restrict__ hbuf,
                                const bf16* __restrict__ wd,
                                const int* __restrict__ counts,
                                const int* __restrict__ tok_list,
                                const float* __restrict__ wts_list,
                                float* __restrict__ out) {
    __shared__ bf16  lA[2][BM * KC];
    __shared__ bf16  lB[2][BN * KC];     // [N][K] transposed
    __shared__ int   toks[BM];
    __shared__ float wtsS[BM];

    int e = blockIdx.z;
    int cnt = counts[e]; if (cnt > CAP) cnt = CAP;
    int mbase = blockIdx.y * BM;
    if (mbase >= cnt) return;
    int nbase = blockIdx.x * BN;
    int tid = threadIdx.x, lane = tid & 31, wid = tid >> 5;

    if (tid < BM) {
        int r = mbase + tid;
        toks[tid] = (r < cnt) ? tok_list[e * CAP + r] : 0;
        wtsS[tid] = (r < cnt) ? wts_list[e * CAP + r] : 0.f;
    }
    __syncthreads();

    const bf16* wdE = wd + (size_t)e * FINT * HID + nbase;

    int waveM = wid >> 2;
    int waveN = wid & 3;
    v8f zero = {0.f,0.f,0.f,0.f,0.f,0.f,0.f,0.f};
    v8f acc[4][2];
#pragma unroll
    for (int i = 0; i < 4; ++i)
#pragma unroll
        for (int j = 0; j < 2; ++j) acc[i][j] = zero;

    int arow = tid >> 1;
    int acol = (tid & 1) * 16;
    const bf16* hrow = hbuf + ((size_t)e * CAP + mbase + arow) * FINT + acol;
    unsigned int aoff[2];
    aoff[0] = (unsigned int)(size_t)&lA[0][arow * KC + acol];
    aoff[1] = (unsigned int)(size_t)&lA[1][arow * KC + acol];

    int lin0 = tid * 8,          kk0 = lin0 >> 7, ff0 = lin0 & 127;
    int lin1 = (tid + 256) * 8,  kk1 = lin1 >> 7, ff1 = lin1 & 127;

#define STAGE_B(K0, BUF)                                                          \
    {                                                                             \
        uint4 vb0 = *(const uint4*)(wdE + (size_t)((K0) + kk0) * HID + ff0);      \
        uint4 vb1 = *(const uint4*)(wdE + (size_t)((K0) + kk1) * HID + ff1);      \
        const bf16* pb0 = (const bf16*)&vb0; const bf16* pb1 = (const bf16*)&vb1; \
        _Pragma("unroll")                                                         \
        for (int j = 0; j < 8; ++j) {                                             \
            lB[BUF][(ff0 + j) * KC + kk0] = pb0[j];                               \
            lB[BUF][(ff1 + j) * KC + kk1] = pb1[j];                               \
        }                                                                         \
    }

    async_ld_b128(aoff[0],      hrow);
    async_ld_b128(aoff[0] + 16, hrow + 8);
    STAGE_B(0, 0)
    wait_async0();
    __syncthreads();

    for (int k0 = 0, it = 0; k0 < FINT; k0 += KC, ++it) {
        int cur = it & 1, nxt = cur ^ 1;
        bool more = (k0 + KC) < FINT;
        if (more) {
            async_ld_b128(aoff[nxt],      hrow + k0 + KC);
            async_ld_b128(aoff[nxt] + 16, hrow + k0 + KC + 8);
            STAGE_B(k0 + KC, nxt)
        }
        int mrow  = lane & 15;
        int khalf = (lane >> 4) * 16;
        v16bf a[4], b[2];
#pragma unroll
        for (int mi = 0; mi < 4; ++mi)
            a[mi] = *(const v16bf*)&lA[cur][(waveM * 64 + mi * 16 + mrow) * KC + khalf];
#pragma unroll
        for (int ni = 0; ni < 2; ++ni)
            b[ni] = *(const v16bf*)&lB[cur][(waveN * 32 + ni * 16 + mrow) * KC + khalf];
#pragma unroll
        for (int mi = 0; mi < 4; ++mi)
#pragma unroll
            for (int ni = 0; ni < 2; ++ni)
                acc[mi][ni] = __builtin_amdgcn_wmma_f32_16x16x32_bf16(
                    false, a[mi], false, b[ni], (short)0, acc[mi][ni], false, false);
        if (more) wait_async0();
        __syncthreads();
    }
#undef STAGE_B

    int ncol  = lane & 15;
    int mhalf = (lane >> 4) * 8;
#pragma unroll
    for (int mi = 0; mi < 4; ++mi)
#pragma unroll
        for (int ni = 0; ni < 2; ++ni)
#pragma unroll
            for (int v = 0; v < 8; ++v) {
                int m = waveM * 64 + mi * 16 + mhalf + v;
                int n = nbase + waveN * 32 + ni * 16 + ncol;
                if (mbase + m < cnt) {
                    float val = wtsS[m] * acc[mi][ni][v];
                    atomicAdd(&out[(size_t)toks[m] * HID + n], val);
                }
            }
}

extern "C" void kernel_launch(void* const* d_in, const int* in_sizes, int n_in,
                              void* d_out, int out_size, void* d_ws, size_t ws_size,
                              hipStream_t stream) {
    const float* x  = (const float*)d_in[0];   // [2,2048,2048]
    const float* gw = (const float*)d_in[1];   // [2048,8]
    const float* wg = (const float*)d_in[2];   // [8,2048,1408]
    const float* wu = (const float*)d_in[3];   // [8,2048,1408]
    const float* wd = (const float*)d_in[4];   // [8,1408,2048]
    float* out = (float*)d_out;                // [4096,2048]

    const long nX = (long)T_TOK * HID;                 // 8,388,608
    const long nW = (long)NEXP * HID * FINT;           // 23,068,672
    char* ws = (char*)d_ws;
    size_t off = 0;
    bf16* xb  = (bf16*)(ws + off); off += (size_t)nX * 2;
    bf16* wgb = (bf16*)(ws + off); off += (size_t)nW * 2;
    bf16* wub = (bf16*)(ws + off); off += (size_t)nW * 2;
    bf16* wdb = (bf16*)(ws + off); off += (size_t)nW * 2;
    bf16* hb  = (bf16*)(ws + off); off += (size_t)NEXP * CAP * FINT * 2;
    int*   counts   = (int*)(ws + off);  off += 256;
    int*   tok_list = (int*)(ws + off);  off += (size_t)NEXP * CAP * 4;
    float* wts_list = (float*)(ws + off);

    // 1) zero output + expert counters
    moe_zero_kernel<<<(int)((nX + 255) / 256), 256, 0, stream>>>(out, nX, counts);

    // 2) fp32 -> bf16 (weights become L2-resident at 138 MB total)
    moe_cvt_bf16_kernel<<<(int)((nX + 1023) / 1024), 256, 0, stream>>>(x,  xb,  nX);
    moe_cvt_bf16_kernel<<<(int)((nW + 1023) / 1024), 256, 0, stream>>>(wg, wgb, nW);
    moe_cvt_bf16_kernel<<<(int)((nW + 1023) / 1024), 256, 0, stream>>>(wu, wub, nW);
    moe_cvt_bf16_kernel<<<(int)((nW + 1023) / 1024), 256, 0, stream>>>(wd, wdb, nW);

    // 3) router: one wave per token
    moe_router_kernel<<<T_TOK / 8, 256, 0, stream>>>(x, gw, counts, tok_list, wts_list);

    // 4) grouped gate+up (SiLU fused)
    dim3 g1(FINT / BN, CAP / BM, NEXP);
    moe_gateup_kernel<<<g1, 256, 0, stream>>>(xb, wgb, wub, counts, tok_list, hb);

    // 5) grouped down + scaled scatter-add
    dim3 g2(HID / BN, CAP / BM, NEXP);
    moe_down_kernel<<<g2, 256, 0, stream>>>(hb, wdb, counts, tok_list, wts_list, out);
}
